// LowRankMultiheadAttention_60679297958127
// MI455X (gfx1250) — matrix operand order repaired and checked by reading, not compiled, not run on hardware
//
#include <hip/hip_runtime.h>
#include <hip/hip_bf16.h>

// LowRankMultiheadAttention for gfx1250 (MI455X).
// HBM-bound on the 512MB attn write; all matmuls via v_wmma_f32_16x16x32_f16.
// attn / final-out stores are non-temporal so the 512MB stream does not evict
// the ~40MB L2-resident operand working set (L2 = 192MB).

#define B_    2
#define S_    2048
#define D_    1024
#define H_    16
#define HD_   64
#define RANK_ 204
#define RP_   224            // RANK padded to multiple of 32
#define BS_   (B_ * S_)      // 4096

typedef __attribute__((ext_vector_type(16))) _Float16 v16h;
typedef __attribute__((ext_vector_type(8)))  _Float16 v8h;
typedef __attribute__((ext_vector_type(8)))  float    v8f;

__device__ __forceinline__ v8f wmma_f16(v16h a, v16h b, v8f c) {
  // (neg_a, A, neg_b, B, c_mod, C, reuse_a, reuse_b)
  return __builtin_amdgcn_wmma_f32_16x16x32_f16(false, a, false, b, (short)0, c,
                                                false, false);
}

// A operand (16x32 f16, MxK). ISA layout: lanes 0-15 hold K={0..7,16..23},
// lanes 16-31 hold K={8..15,24..31}, row M = lane&15.
__device__ __forceinline__ v16h load_A_f16(const _Float16* base, int ld) {
  const int lane = threadIdx.x & 31;
  const int m = lane & 15, hi = lane >> 4;
  const _Float16* r = base + (size_t)m * ld + hi * 8;
  v8h lo = *(const v8h*)(r);
  v8h hh = *(const v8h*)(r + 16);
  v16h out;
#pragma unroll
  for (int i = 0; i < 8; ++i) { out[i] = lo[i]; out[i + 8] = hh[i]; }
  return out;
}

// Same, sourcing f32 and converting (saves a full f32->f16 staging pass).
__device__ __forceinline__ v16h load_A_f32(const float* base, int ld) {
  const int lane = threadIdx.x & 31;
  const int m = lane & 15, hi = lane >> 4;
  const float* r = base + (size_t)m * ld + hi * 8;
  v16h out;
#pragma unroll
  for (int i = 0; i < 8; ++i) {
    out[i]     = (_Float16)r[i];
    out[i + 8] = (_Float16)r[16 + i];
  }
  return out;
}

// B operand (32x16 f16, KxN) loaded from B^T stored row-major [N][K]:
// lane holds column n = lane&15, K={0..7,16..23} (lanes 0-15) / {8..15,24..31}.
__device__ __forceinline__ v16h load_B_f16(const _Float16* baseT, int ld) {
  const int lane = threadIdx.x & 31;
  const int n = lane & 15, hi = lane >> 4;
  const _Float16* r = baseT + (size_t)n * ld + hi * 8;
  v8h lo = *(const v8h*)(r);
  v8h hh = *(const v8h*)(r + 16);
  v16h out;
#pragma unroll
  for (int i = 0; i < 8; ++i) { out[i] = lo[i]; out[i + 8] = hh[i]; }
  return out;
}

// ---------------------------------------------------------------- prep weights
// V [204,1024] f32 -> Vh [224,1024] f16 (zero rows >=204)
// U [1024,204] f32 -> Uh [1024,224] f16 (zero cols >=204)
__global__ void prep_weights(const float* __restrict__ Vw,
                             const float* __restrict__ Uw,
                             _Float16* __restrict__ Vh,
                             _Float16* __restrict__ Uh) {
  const int N = RP_ * D_;
  for (int i = blockIdx.x * blockDim.x + threadIdx.x; i < N;
       i += gridDim.x * blockDim.x) {
    int r = i / D_, c = i - r * D_;
    Vh[i] = (r < RANK_) ? (_Float16)Vw[r * D_ + c] : (_Float16)0.f;
    int r2 = i / RP_, c2 = i - r2 * RP_;
    Uh[i] = (c2 < RANK_) ? (_Float16)Uw[r2 * RANK_ + c2] : (_Float16)0.f;
  }
}

// ---------------------------------------------------------------- GEMM1
// T[4096,224] = X[4096,1024] @ V^T   (V stored padded [224,1024] = B^T)
// one wave per 16x16 tile; grid (256, 14), block 32. Templated A source so
// the K-loop is a pure load->wmma stream (no per-iteration branch).
template <bool USE_F16>
__global__ void gemm1_lowrank(const float* __restrict__ A32,
                              const _Float16* __restrict__ A16,
                              const _Float16* __restrict__ Vh,
                              _Float16* __restrict__ T) {
  const int mt = blockIdx.x, nt = blockIdx.y;
  const _Float16* bbase = Vh + (size_t)nt * 16 * D_;
  v8f c = {};
  for (int k0 = 0; k0 < D_; k0 += 32) {
    if (k0 + 32 < D_) {  // prefetch next K-step (global_prefetch_b8)
      __builtin_prefetch(bbase + k0 + 32, 0, 1);
      if (USE_F16)
        __builtin_prefetch(A16 + (size_t)mt * 16 * D_ + k0 + 32, 0, 1);
      else
        __builtin_prefetch(A32 + (size_t)mt * 16 * D_ + k0 + 32, 0, 1);
    }
    v16h a = USE_F16 ? load_A_f16(A16 + (size_t)mt * 16 * D_ + k0, D_)
                     : load_A_f32(A32 + (size_t)mt * 16 * D_ + k0, D_);
    v16h b = load_B_f16(bbase + k0, D_);
    c = wmma_f16(a, b, c);
  }
  const int lane = threadIdx.x & 31;
  const int n = lane & 15, hi = lane >> 4;
#pragma unroll
  for (int r = 0; r < 8; ++r) {
    int m = hi * 8 + r;
    T[(size_t)(mt * 16 + m) * RP_ + nt * 16 + n] = (_Float16)c[r];
  }
}

// ---------------------------------------------------------------- GEMM2
// Y[4096,1024] = T[4096,224] @ U^T + bias  (U padded [1024,224] = B^T)
// one wave per 16x64 tile (A reuse x4); grid (256, 16), block 32.
// mode 0: f16 head-split [B,H,S,64]   (q, k)
// mode 1: f16 transposed [B,H,64,S]   (v)
// mode 2: f32 flat [B,S,D]            (final output, non-temporal)
__global__ void gemm2_lowrank(const _Float16* __restrict__ T,
                              const _Float16* __restrict__ Uh,
                              const float* __restrict__ bias,
                              void* __restrict__ dst, int mode) {
  const int mt = blockIdx.x, bn = blockIdx.y;
  v8f acc[4] = {{}, {}, {}, {}};
  for (int k0 = 0; k0 < RP_; k0 += 32) {
    if (k0 + 32 < RP_)
      __builtin_prefetch(T + (size_t)mt * 16 * RP_ + k0 + 32, 0, 1);
    v16h a = load_A_f16(T + (size_t)mt * 16 * RP_ + k0, RP_);
#pragma unroll
    for (int t = 0; t < 4; ++t) {
      v16h b = load_B_f16(Uh + (size_t)(bn * 64 + t * 16) * RP_ + k0, RP_);
      acc[t] = wmma_f16(a, b, acc[t]);
    }
  }
  const int lane = threadIdx.x & 31;
  const int nl = lane & 15, hi = lane >> 4;
#pragma unroll
  for (int t = 0; t < 4; ++t) {
    int n = bn * 64 + t * 16 + nl;
    float bv = bias[n];
    int h = n >> 6, hd = n & 63;
#pragma unroll
    for (int r = 0; r < 8; ++r) {
      int row = mt * 16 + hi * 8 + r;
      int bb = row >> 11, s = row & (S_ - 1);
      float y = acc[t][r] + bv;
      if (mode == 0) {
        ((_Float16*)dst)[(((size_t)bb * H_ + h) * S_ + s) * HD_ + hd] =
            (_Float16)y;
      } else if (mode == 1) {
        ((_Float16*)dst)[(((size_t)bb * H_ + h) * HD_ + hd) * S_ + s] =
            (_Float16)y;
      } else {
        __builtin_nontemporal_store(y, (float*)dst + (size_t)row * D_ + n);
      }
    }
  }
}

// ---------------------------------------------------------------- attention
// One 8-wave block per (b*h, 16-row q tile). 16x2048 f32 score strip lives in
// LDS (128KB of the 320KB WGP LDS); attn is written to HBM exactly once, with
// non-temporal stores (512MB stream > 192MB L2 -> keep L2 for reused tensors).
// grid (S/16, B*H), block 256, dynamic LDS = 128KB scores + 64KB f16 probs.
__global__ void attn_kernel(const _Float16* __restrict__ qh,
                            const _Float16* __restrict__ kh,
                            const _Float16* __restrict__ vhT,
                            float* __restrict__ attn_out,
                            _Float16* __restrict__ outm) {
  extern __shared__ char smem[];
  float* sc = (float*)smem;                              // [16][2048] f32
  _Float16* pr = (_Float16*)(smem + 16 * S_ * 4);        // [16][2048] f16
  float* part = sc;                                      // reuse for partials

  const int qt = blockIdx.x, bh = blockIdx.y;
  const int w = threadIdx.x >> 5, lane = threadIdx.x & 31;
  const int qbase = qt * 16;
  const int nl = lane & 15, hi = lane >> 4;

  // ---- phase 1: scores = (Q K^T) * 1/sqrt(64), causal masked, into LDS
  const _Float16* qptr = qh + ((size_t)bh * S_ + qbase) * HD_;
  v16h a0 = load_A_f16(qptr + 0, HD_);
  v16h a1 = load_A_f16(qptr + 32, HD_);
#pragma unroll 1
  for (int nt = 0; nt < 16; ++nt) {
    int col0 = w * 256 + nt * 16;
    if (col0 <= qbase + 15) {
      const _Float16* kp = kh + ((size_t)bh * S_ + col0) * HD_;
      v16h b0 = load_B_f16(kp + 0, HD_);
      v16h b1 = load_B_f16(kp + 32, HD_);
      v8f c = {};
      c = wmma_f16(a0, b0, c);
      c = wmma_f16(a1, b1, c);
#pragma unroll
      for (int r = 0; r < 8; ++r) {
        int m = hi * 8 + r;
        int col = col0 + nl;
        sc[m * S_ + col] =
            (col <= qbase + m) ? c[r] * 0.125f : -__builtin_inff();
      }
    } else {
#pragma unroll
      for (int r = 0; r < 8; ++r)
        sc[(hi * 8 + r) * S_ + col0 + nl] = -__builtin_inff();
    }
  }
  __syncthreads();

  // ---- phase 2: row softmax (wave w owns rows 2w, 2w+1); single attn write
#pragma unroll 1
  for (int rr = 0; rr < 2; ++rr) {
    int m = w * 2 + rr;
    float mx = -__builtin_inff();
    for (int j = lane; j < S_; j += 32) mx = fmaxf(mx, sc[m * S_ + j]);
#pragma unroll
    for (int off = 16; off > 0; off >>= 1) mx = fmaxf(mx, __shfl_xor(mx, off));
    float sum = 0.f;
    for (int j = lane; j < S_; j += 32) sum += __expf(sc[m * S_ + j] - mx);
#pragma unroll
    for (int off = 16; off > 0; off >>= 1) sum += __shfl_xor(sum, off);
    float inv = 1.f / sum;
    float* arow = attn_out + ((size_t)bh * S_ + qbase + m) * S_;
    for (int j = lane; j < S_; j += 32) {
      float p = __expf(sc[m * S_ + j] - mx) * inv;
      pr[m * S_ + j] = (_Float16)p;
      __builtin_nontemporal_store(p, arow + j);   // TH=NT streaming store
    }
  }
  __syncthreads();

  // ---- phase 3: out tile = P @ V (per-wave K split, LDS reduction)
  v8f acc[4] = {{}, {}, {}, {}};
#pragma unroll 1
  for (int ks = 0; ks < 8; ++ks) {
    int kb = w * 256 + ks * 32;
    if (kb > qbase + 15) break;  // fully-masked K range -> probs are zero
    v16h a = load_A_f16(pr + kb, S_);
#pragma unroll
    for (int nt = 0; nt < 4; ++nt) {
      const _Float16* vp = vhT + ((size_t)bh * HD_ + nt * 16) * S_ + kb;
      v16h b = load_B_f16(vp, S_);
      acc[nt] = wmma_f16(a, b, acc[nt]);
    }
  }
#pragma unroll
  for (int nt = 0; nt < 4; ++nt)
#pragma unroll
    for (int r = 0; r < 8; ++r)
      part[w * 1024 + (hi * 8 + r) * 64 + nt * 16 + nl] = acc[nt][r];
  __syncthreads();

  const int h = bh & (H_ - 1), bb = bh >> 4;
  for (int idx = threadIdx.x; idx < 1024; idx += 256) {
    float s2 = 0.f;
#pragma unroll
    for (int ww = 0; ww < 8; ++ww) s2 += part[ww * 1024 + idx];
    int m = idx >> 6, d = idx & 63;
    outm[((size_t)bb * S_ + qbase + m) * D_ + h * HD_ + d] = (_Float16)s2;
  }
}

// ---------------------------------------------------------------- launch
extern "C" void kernel_launch(void* const* d_in, const int* in_sizes, int n_in,
                              void* d_out, int out_size, void* d_ws,
                              size_t ws_size, hipStream_t stream) {
  (void)in_sizes; (void)n_in; (void)out_size; (void)ws_size;
  const float* q  = (const float*)d_in[0];
  const float* k  = (const float*)d_in[1];
  const float* v  = (const float*)d_in[2];
  const float* Vw[4] = {(const float*)d_in[3], (const float*)d_in[6],
                        (const float*)d_in[9], (const float*)d_in[12]};
  const float* Uw[4] = {(const float*)d_in[4], (const float*)d_in[7],
                        (const float*)d_in[10], (const float*)d_in[13]};
  const float* bw[4] = {(const float*)d_in[5], (const float*)d_in[8],
                        (const float*)d_in[11], (const float*)d_in[14]};

  float* out_final = (float*)d_out;                       // [B,S,D]
  float* attn_out = (float*)d_out + (size_t)BS_ * D_;     // [B*H,S,S]

  // workspace carve-up (256B aligned)
  char* ws = (char*)d_ws;
  size_t off = 0;
  auto alloc = [&](size_t bytes) {
    void* p = ws + off;
    off = (off + bytes + 255) & ~(size_t)255;
    return p;
  };
  _Float16* qh   = (_Float16*)alloc((size_t)BS_ * D_ * 2);   // [B,H,S,64]
  _Float16* khd  = (_Float16*)alloc((size_t)BS_ * D_ * 2);   // [B,H,S,64]
  _Float16* vhT  = (_Float16*)alloc((size_t)BS_ * D_ * 2);   // [B,H,64,S]
  _Float16* outm = (_Float16*)alloc((size_t)BS_ * D_ * 2);   // [B,S,D]
  _Float16* T    = (_Float16*)alloc((size_t)BS_ * RP_ * 2);  // [4096,224]
  _Float16* Vh[4];
  _Float16* Uh[4];
  for (int i = 0; i < 4; ++i) {
    Vh[i] = (_Float16*)alloc((size_t)RP_ * D_ * 2);
    Uh[i] = (_Float16*)alloc((size_t)RP_ * D_ * 2);
  }

  // pad+convert all weights
  for (int i = 0; i < 4; ++i)
    prep_weights<<<dim3(448), dim3(256), 0, stream>>>(Vw[i], Uw[i], Vh[i],
                                                      Uh[i]);

  // input projections q -> qh (mode0), k -> kh (mode0), v -> vhT (mode1)
  const float* xin[3] = {q, k, v};
  void* xdst[3] = {(void*)qh, (void*)khd, (void*)vhT};
  int xmode[3] = {0, 0, 1};
  for (int p = 0; p < 3; ++p) {
    gemm1_lowrank<false><<<dim3(BS_ / 16, RP_ / 16), dim3(32), 0, stream>>>(
        xin[p], (const _Float16*)nullptr, Vh[p], T);
    gemm2_lowrank<<<dim3(BS_ / 16, D_ / 64), dim3(32), 0, stream>>>(
        T, Uh[p], bw[p], xdst[p], xmode[p]);
  }

  // attention: 192KB dynamic LDS per workgroup (scores f32 + probs f16)
  const int lds_bytes = 16 * S_ * 4 + 16 * S_ * 2;
  (void)hipFuncSetAttribute((const void*)attn_kernel,
                            hipFuncAttributeMaxDynamicSharedMemorySize,
                            lds_bytes);
  attn_kernel<<<dim3(S_ / 16, B_ * H_), dim3(256), lds_bytes, stream>>>(
      qh, khd, vhT, attn_out, outm);

  // output projection -> f32 d_out
  gemm1_lowrank<true><<<dim3(BS_ / 16, RP_ / 16), dim3(32), 0, stream>>>(
      (const float*)nullptr, outm, Vh[3], T);
  gemm2_lowrank<<<dim3(BS_ / 16, D_ / 64), dim3(32), 0, stream>>>(
      T, Uh[3], bw[3], (void*)out_final, 2);
}
